// MultiHeadAttention_29248727286442
// MI455X (gfx1250) — matrix operand order repaired and checked by reading, not compiled, not run on hardware
//
#include <hip/hip_runtime.h>
#include <stdint.h>

typedef __attribute__((ext_vector_type(16))) __bf16 v16bf;
typedef __attribute__((ext_vector_type(8)))  float  v8f;

union Frag {
  v16bf v;
  uint4 q[2];
};

__device__ __forceinline__ unsigned short f2bf(float f) {
  uint32_t x = __float_as_uint(f);
  x += 0x7FFFu + ((x >> 16) & 1u);   // round-to-nearest-even
  return (unsigned short)(x >> 16);
}

// xor-lane reduction step via ds_swizzle (group-of-32 mode, and_mask=0x1F).
// XOR masks <= 8 stay inside each 16-lane half (matches WMMA C-row layout).
template <int PAT>
__device__ __forceinline__ float swz(float x) {
  return __int_as_float(__builtin_amdgcn_ds_swizzle(__float_as_int(x), PAT));
}
#define SWZ_X1 0x041F
#define SWZ_X2 0x081F
#define SWZ_X4 0x101F
#define SWZ_X8 0x201F

#define WMMA_BF16(A, B, C) \
  __builtin_amdgcn_wmma_f32_16x16x32_bf16(false, (A), false, (B), (short)0, (C), false, false)

// ---------------------------------------------------------------------------
// GEMM + bias:  C[M,N] = (A[M,K] @ W[K,N] + bias) * cscale
//   A: fp32 or bf16 (template), W/bias fp32.
//   OMODE 0: C bf16 row-major (ws)          [Q (cscale=0.125*log2e), K proj]
//   OMODE 1: C fp32 row-major (d_out)       [output projection]
//   OMODE 2: C bf16 V^T-per-head layout     [V projection -> attention B-matrix]
// Block: 128x128 tile, 256 threads (8 waves), each wave 32M x 64N, Kc=64.
// LDS row stride 72 elems = 144 B (multiple of 16 B -> aligned ds_load_b128).
// ---------------------------------------------------------------------------
template <bool A_BF16, int OMODE>
__global__ void __launch_bounds__(256) gemm_bias_kernel(
    const void* __restrict__ Aptr, const float* __restrict__ W,
    const float* __restrict__ bias, void* __restrict__ Cptr,
    int M, int N, int K, float cscale) {
  __shared__ unsigned short Asub[128][72];
  __shared__ unsigned short Bsub[128][72];

  const int tid = threadIdx.x;
  const int blockM = blockIdx.y * 128;
  const int blockN = blockIdx.x * 128;
  const int wave = tid >> 5;
  const int lane = tid & 31;
  const int hf = lane >> 4;
  const int ln = lane & 15;
  const int waveM = (wave & 3) * 32;
  const int waveN = (wave >> 2) * 64;

  const float* Af = (const float*)Aptr;
  const unsigned short* Ab = (const unsigned short*)Aptr;

  v8f acc[2][4] = {};

  for (int kk = 0; kk < K; kk += 64) {
    // ---- stage A tile (128 x 64); packed bf16 pair stores ----
#pragma unroll
    for (int i = 0; i < 8; ++i) {
      int qi = tid + i * 256;          // 2048 groups of 4 elements
      int r = qi >> 4;
      int c = (qi & 15) << 2;
      long g = (long)(blockM + r) * K + kk + c;
      if (A_BF16) {
        *(uint2*)&Asub[r][c] = *(const uint2*)(Ab + g);
      } else {
        float4 d = *(const float4*)(Af + g);
        uint32_t p0 = (uint32_t)f2bf(d.x) | ((uint32_t)f2bf(d.y) << 16);
        uint32_t p1 = (uint32_t)f2bf(d.z) | ((uint32_t)f2bf(d.w) << 16);
        *(uint2*)&Asub[r][c] = make_uint2(p0, p1);
      }
    }
    // ---- stage W tile transposed: Bsub[n][k] ----
#pragma unroll
    for (int i = 0; i < 8; ++i) {
      int qi = tid + i * 256;
      int kr = qi >> 5;                // 0..63
      int c = (qi & 31) << 2;          // 0..124
      float4 d = *(const float4*)(W + (long)(kk + kr) * N + blockN + c);
      Bsub[c + 0][kr] = f2bf(d.x);
      Bsub[c + 1][kr] = f2bf(d.y);
      Bsub[c + 2][kr] = f2bf(d.z);
      Bsub[c + 3][kr] = f2bf(d.w);
    }
    __syncthreads();

#pragma unroll
    for (int ks = 0; ks < 64; ks += 32) {
      Frag a[2], b[4];
#pragma unroll
      for (int tm = 0; tm < 2; ++tm) {
        int m = waveM + tm * 16 + ln;
        a[tm].q[0] = *(const uint4*)&Asub[m][ks + hf * 8];
        a[tm].q[1] = *(const uint4*)&Asub[m][ks + 16 + hf * 8];
      }
#pragma unroll
      for (int tn = 0; tn < 4; ++tn) {
        int n = waveN + tn * 16 + ln;
        b[tn].q[0] = *(const uint4*)&Bsub[n][ks + hf * 8];
        b[tn].q[1] = *(const uint4*)&Bsub[n][ks + 16 + hf * 8];
      }
#pragma unroll
      for (int tm = 0; tm < 2; ++tm)
#pragma unroll
        for (int tn = 0; tn < 4; ++tn)
          acc[tm][tn] = WMMA_BF16(a[tm].v, b[tn].v, acc[tm][tn]);
    }
    __syncthreads();
  }

  // ---- epilogue ----
#pragma unroll
  for (int tm = 0; tm < 2; ++tm) {
#pragma unroll
    for (int tn = 0; tn < 4; ++tn) {
      int gcol = blockN + waveN + tn * 16 + ln;
      float bb = bias[gcol];
#pragma unroll
      for (int r = 0; r < 8; ++r) {
        int grow = blockM + waveM + tm * 16 + r + 8 * hf;
        float val = (acc[tm][tn][r] + bb) * cscale;
        if (OMODE == 1) {
          ((float*)Cptr)[(long)grow * N + gcol] = val;
        } else if (OMODE == 0) {
          ((unsigned short*)Cptr)[(long)grow * N + gcol] = f2bf(val);
        } else {
          // V^T per head (hardcodes B=2, S=2048, H=1024, nh=16, dk=64):
          // flat elem fb within batch -> head hbl, row rr, dk dd; store [dd][rr]
          int b = grow >> 11;
          int fb = ((grow & 2047) << 10) | gcol;
          int hbl = fb >> 17;
          int rem = fb & 131071;
          long dst = ((long)((b << 4) + hbl) << 17) +
                     ((long)(rem & 63) << 11) + (rem >> 6);
          ((unsigned short*)Cptr)[dst] = f2bf(val);
        }
      }
    }
  }
}

// ---------------------------------------------------------------------------
// Flash attention over contiguous per-head 2048x64 Q/K (bf16) and per-head
// V^T [64][2048] (bf16). Q is pre-scaled by 0.125*log2(e) so softmax uses
// exp2 directly. Grid: (S/64 q-tiles, B*nh). Block: 128 thr = 4 waves, each
// wave owns 16 query rows. K-tile and V^T-tile are staged into LDS with async
// global->LDS copies (double-buffered; next tile's copy overlaps compute) and
// shared by all 4 waves.
// ---------------------------------------------------------------------------
__global__ void __launch_bounds__(128) flash_attn_kernel(
    const unsigned short* __restrict__ Q, const unsigned short* __restrict__ Km,
    const unsigned short* __restrict__ Vt, unsigned short* __restrict__ O) {
  const int SH = 2048, DK = 64;
  __shared__ unsigned short Klds[2][64 * 64];   // [key][dk], flat copy
  __shared__ unsigned short Vlds[2][64 * 64];   // [dk][key]
  __shared__ unsigned short Plds[4][16][72];    // per-wave P tile [m][key]

  const int hb = blockIdx.y;                    // b*16 + head
  const long base = (long)hb * SH * DK;
  const unsigned short* Qh = Q + base;
  const unsigned short* Kh = Km + base;
  const unsigned short* Vth = Vt + base;        // [dk][2048]
  unsigned short* Oh = O + base;

  const int tid = threadIdx.x;
  const int wave = tid >> 5;
  const int lane = tid & 31;
  const int hf = lane >> 4;
  const int ln = lane & 15;
  const int qRow0 = blockIdx.x * 64 + wave * 16;

  // Async-copy one 64-key tile of K (8KB contiguous) and V^T (64 x 128B rows)
  // into LDS buffer `buf`: 512 16B chunks each, 4 per thread per matrix.
  auto issue_tile = [&](int buf, int kt2) {
    unsigned kbase = (unsigned)(uintptr_t)&Klds[buf][0];
    unsigned vbase = (unsigned)(uintptr_t)&Vlds[buf][0];
#pragma unroll
    for (int i = 0; i < 4; ++i) {
      int c = tid + i * 128;                       // chunk 0..511
      const unsigned short* kg = Kh + (long)kt2 * DK + c * 8;
      unsigned kl = kbase + c * 16;
      asm volatile("global_load_async_to_lds_b128 %0, %1, off"
                   :: "v"(kl), "v"(kg) : "memory");
      int d = c >> 3;
      int off = (c & 7) << 3;
      const unsigned short* vg = Vth + (long)d * SH + kt2 + off;
      unsigned vl = vbase + c * 16;
      asm volatile("global_load_async_to_lds_b128 %0, %1, off"
                   :: "v"(vl), "v"(vg) : "memory");
    }
  };

  // Q fragments (A matrix, d_k = 64 -> two K-steps), loaded once from global.
  Frag aQ[2];
  {
    const unsigned short* qr = Qh + (long)(qRow0 + ln) * DK;
#pragma unroll
    for (int s = 0; s < 2; ++s) {
      aQ[s].q[0] = *(const uint4*)(qr + s * 32 + hf * 8);
      aQ[s].q[1] = *(const uint4*)(qr + s * 32 + 16 + hf * 8);
    }
  }

  v8f o[4] = {};
  float mrow[8], lrow[8];
#pragma unroll
  for (int r = 0; r < 8; ++r) { mrow[r] = -3.0e38f; lrow[r] = 0.f; }

  issue_tile(0, 0);
  int cur = 0;

  for (int kt = 0; kt < SH; kt += 64) {
    asm volatile("s_wait_asynccnt 0" ::: "memory");
    __syncthreads();                      // staged tile visible to all waves
    if (kt + 64 < SH) issue_tile(cur ^ 1, kt + 64);   // overlap next copy

    const unsigned short* Kl = &Klds[cur][0];
    const unsigned short* Vl = &Vlds[cur][0];

    // ---- scores = Qs @ K^T : 16 rows x 64 keys per wave ----
    v8f sc[4] = {};
#pragma unroll
    for (int s = 0; s < 2; ++s) {
      Frag bK[4];
#pragma unroll
      for (int t = 0; t < 4; ++t) {
        const unsigned short* kr = Kl + (t * 16 + ln) * 64 + s * 32 + hf * 8;
        bK[t].q[0] = *(const uint4*)kr;
        bK[t].q[1] = *(const uint4*)(kr + 16);
      }
#pragma unroll
      for (int t = 0; t < 4; ++t)
        sc[t] = WMMA_BF16(aQ[s].v, bK[t].v, sc[t]);
    }

    // ---- online softmax in base-2; row M = r + 8*hf in a 16-lane half ----
#pragma unroll
    for (int r = 0; r < 8; ++r) {
      float mx = fmaxf(fmaxf(sc[0][r], sc[1][r]), fmaxf(sc[2][r], sc[3][r]));
      mx = fmaxf(mx, swz<SWZ_X1>(mx));
      mx = fmaxf(mx, swz<SWZ_X2>(mx));
      mx = fmaxf(mx, swz<SWZ_X4>(mx));
      mx = fmaxf(mx, swz<SWZ_X8>(mx));
      float mnew = fmaxf(mrow[r], mx);
      float alpha = exp2f(mrow[r] - mnew);
      mrow[r] = mnew;
      float rs = 0.f;
#pragma unroll
      for (int t = 0; t < 4; ++t) {
        float p = exp2f(sc[t][r] - mnew);
        sc[t][r] = p;
        rs += p;
      }
      rs += swz<SWZ_X1>(rs);
      rs += swz<SWZ_X2>(rs);
      rs += swz<SWZ_X4>(rs);
      rs += swz<SWZ_X8>(rs);
      lrow[r] = lrow[r] * alpha + rs;
#pragma unroll
      for (int t = 0; t < 4; ++t) o[t][r] *= alpha;
    }

    // ---- C-layout -> A-layout reshape of P via (wave-local) LDS ----
#pragma unroll
    for (int t = 0; t < 4; ++t)
#pragma unroll
      for (int r = 0; r < 8; ++r)
        Plds[wave][r + 8 * hf][t * 16 + ln] = f2bf(sc[t][r]);

    // ---- O += P @ V  (B-fragments from staged V^T) ----
#pragma unroll
    for (int s = 0; s < 2; ++s) {
      Frag aP, bV[4];
      aP.q[0] = *(const uint4*)&Plds[wave][ln][s * 32 + hf * 8];
      aP.q[1] = *(const uint4*)&Plds[wave][ln][s * 32 + 16 + hf * 8];
#pragma unroll
      for (int t = 0; t < 4; ++t) {
        const unsigned short* vr = Vl + (t * 16 + ln) * 64 + s * 32 + hf * 8;
        bV[t].q[0] = *(const uint4*)vr;
        bV[t].q[1] = *(const uint4*)(vr + 16);
      }
#pragma unroll
      for (int t = 0; t < 4; ++t)
        o[t] = WMMA_BF16(aP.v, bV[t].v, o[t]);
    }
    __syncthreads();   // all waves done with buffer `cur` before reuse
    cur ^= 1;
  }

  // ---- normalize and store (flat head layout == reference reshape) ----
#pragma unroll
  for (int r = 0; r < 8; ++r) {
    float inv = 1.0f / lrow[r];
    int row = qRow0 + r + 8 * hf;
#pragma unroll
    for (int t = 0; t < 4; ++t)
      Oh[(long)row * DK + t * 16 + ln] = f2bf(o[t][r] * inv);
  }
}

// ---------------------------------------------------------------------------
extern "C" void kernel_launch(void* const* d_in, const int* in_sizes, int n_in,
                              void* d_out, int out_size, void* d_ws, size_t ws_size,
                              hipStream_t stream) {
  (void)in_sizes; (void)n_in; (void)out_size; (void)ws_size;

  const float* query = (const float*)d_in[0];
  const float* key   = (const float*)d_in[1];
  const float* value = (const float*)d_in[2];
  const float* Wq = (const float*)d_in[3];
  const float* bq = (const float*)d_in[4];
  const float* Wk = (const float*)d_in[5];
  const float* bk = (const float*)d_in[6];
  const float* Wv = (const float*)d_in[7];
  const float* bv = (const float*)d_in[8];
  const float* Wo = (const float*)d_in[9];
  const float* bo = (const float*)d_in[10];
  float* out = (float*)d_out;

  const int M = 4096, N = 1024, K = 1024;   // M = B*S
  size_t qsz = (size_t)M * N;
  unsigned short* Qb  = (unsigned short*)d_ws;
  unsigned short* Kb  = Qb + qsz;
  unsigned short* Vtb = Kb + qsz;           // per-head transposed V
  unsigned short* Ab  = Vtb + qsz;          // attention output (bf16)

  // Fold softmax scaling (1/sqrt(dk) = 0.125) and log2(e) into Q so the
  // attention kernel uses exp2 with no per-score multiply.
  const float qscale = 0.125f * 1.44269504088896340736f;

  dim3 gg(N / 128, M / 128);
  gemm_bias_kernel<false, 0><<<gg, 256, 0, stream>>>(query, Wq, bq, Qb,  M, N, K, qscale);
  gemm_bias_kernel<false, 0><<<gg, 256, 0, stream>>>(key,   Wk, bk, Kb,  M, N, K, 1.0f);
  gemm_bias_kernel<false, 2><<<gg, 256, 0, stream>>>(value, Wv, bv, Vtb, M, N, K, 1.0f);

  flash_attn_kernel<<<dim3(2048 / 64, 32), 128, 0, stream>>>(Qb, Kb, Vtb, Ab);

  gemm_bias_kernel<true, 1><<<gg, 256, 0, stream>>>(Ab, Wo, bo, out, M, N, K, 1.0f);
}
